// Decoder_72438918414350
// MI455X (gfx1250) — compile-verified, matrix operand order checked by reading, not run on hardware
//
#include <hip/hip_runtime.h>
#include <hip/hip_bf16.h>

// ---------------------------------------------------------------------------
// Types for CDNA5 WMMA (wave32): v_wmma_f32_16x16x32_bf16
// ---------------------------------------------------------------------------
typedef __attribute__((ext_vector_type(16))) __bf16 v16bf;
typedef __attribute__((ext_vector_type(8)))  float  v8f;

// fp32 -> bf16 with round-to-nearest-even
__device__ __forceinline__ unsigned short f2bf(float x) {
    union { float f; unsigned u; } a; a.f = x;
    unsigned r = a.u + 0x7FFFu + ((a.u >> 16) & 1u);
    return (unsigned short)(r >> 16);
}

// Load a 16x32 bf16 fragment (A layout; B uses the same per-lane mapping with
// row == local N because we store B transposed [n][k] in LDS).
// ISA 7.12.2: lane L holds M = L&15; K-halves: lanes<16 -> K base 0, lanes>=16
// -> K base 8; elements 0..7 = K {kb..kb+7}, elements 8..15 = K {kb+16..kb+23}.
// LDS row stride = 34 halves (17 dwords) -> conflict-free.
__device__ __forceinline__ v16bf load_frag(const unsigned short* lds,
                                           int row_base, int lane) {
    union { v16bf v; unsigned u[8]; } f;
    const int row = row_base + (lane & 15);
    const unsigned* p = (const unsigned*)(lds + row * 34);
    const int kb = (lane >> 4) << 3;               // 0 or 8
#pragma unroll
    for (int j = 0; j < 8; ++j) {
        int k2 = kb + ((j & 3) << 1) + ((j >> 2) << 4);  // even K index
        f.u[j] = p[k2 >> 1];
    }
    return f.v;
}

// Per-thread register staging of one 128x32 A tile + 32x64 B tile.
struct StageRegs {
    float4 a[4];
    float  b[8];
};

__device__ __forceinline__ void stage_global(const float* __restrict__ A, int lda,
                                             const float* __restrict__ B, int ldb,
                                             int m0, int n0, int N, int k0,
                                             int tid, StageRegs& r) {
    // A tile: 128 rows x 32 cols, 4 float4 per thread (coalesced, 8 thr/row)
#pragma unroll
    for (int i = 0; i < 4; ++i) {
        int slot = tid + (i << 8);         // 0..1023
        int row  = slot >> 3;              // 0..127
        int c4   = (slot & 7) << 2;        // 0..28
        r.a[i] = *(const float4*)&A[(size_t)(m0 + row) * lda + k0 + c4];
    }
    // B tile: 32 rows x 64 cols, 8 scalars per thread (coalesced over n)
    int nl = tid & 63;
    int kf = tid >> 6;
    int gc = n0 + nl;
    bool ok = (gc < N);
#pragma unroll
    for (int j = 0; j < 8; ++j) {
        int kk = kf + (j << 2);
        r.b[j] = ok ? B[(size_t)(k0 + kk) * ldb + gc] : 0.0f;
    }
}

__device__ __forceinline__ void stage_store(unsigned short* __restrict__ sa,
                                            unsigned short* __restrict__ sb,
                                            int tid, const StageRegs& r) {
#pragma unroll
    for (int i = 0; i < 4; ++i) {
        int slot = tid + (i << 8);
        int row  = slot >> 3;
        int c4   = (slot & 7) << 2;
        unsigned short* d = &sa[row * 34 + c4];
        d[0] = f2bf(r.a[i].x); d[1] = f2bf(r.a[i].y);
        d[2] = f2bf(r.a[i].z); d[3] = f2bf(r.a[i].w);
    }
    int nl = tid & 63;
    int kf = tid >> 6;
#pragma unroll
    for (int j = 0; j < 8; ++j) {
        int kk = kf + (j << 2);
        sb[nl * 34 + kk] = f2bf(r.b[j]);       // transposed [n][k]
    }
}

// ---------------------------------------------------------------------------
// Generic GEMM: C[M,N] = A[M,K] @ B[K,N] (+bias[n]) (+= existing C if ACC)
// fp32 in / fp32 out, bf16 WMMA compute with f32 accumulation.
// Requirements: M % 128 == 0, K % 32 == 0. N arbitrary (guarded).
// Block: 256 threads (8 waves). Block tile 128x64, wave tile 32x32.
// Double-buffered LDS; next tile's global loads issued before current WMMAs.
// ---------------------------------------------------------------------------
__global__ __launch_bounds__(256)
void gemm_bf16_wmma(const float* __restrict__ A, int lda,
                    const float* __restrict__ B, int ldb,
                    const float* __restrict__ bias,
                    float* __restrict__ C, int ldc,
                    int M, int N, int K, int acc_flag) {
    __shared__ unsigned short s_a[2][128 * 34];   // [m][k]  2 x 8704 B
    __shared__ unsigned short s_b[2][64 * 34];    // [n][k]  2 x 4352 B

    const int tid  = threadIdx.x;
    const int wave = tid >> 5;
    const int lane = tid & 31;
    const int wm   = (wave & 3) << 5;          // 0,32,64,96
    const int wn   = (wave >> 2) << 5;         // 0,32
    const int m0   = blockIdx.y << 7;          // 128-row block
    const int n0   = blockIdx.x << 6;          // 64-col block

    v8f acc00 = {}, acc01 = {}, acc10 = {}, acc11 = {};

    const int nk = K >> 5;
    StageRegs rg;

    // prologue: stage tile 0 into buffer 0
    stage_global(A, lda, B, ldb, m0, n0, N, 0, tid, rg);
    stage_store(s_a[0], s_b[0], tid, rg);

    for (int i = 0; i < nk; ++i) {
        __syncthreads();   // buf[cur] ready; all reads of buf[nxt] (iter i-1) done

        const int cur = i & 1;
        const int nxt = cur ^ 1;
        const bool more = (i + 1) < nk;

        // issue next tile's global loads early (overlap with WMMA below)
        if (more)
            stage_global(A, lda, B, ldb, m0, n0, N, (i + 1) << 5, tid, rg);

        // WMMA compute on current buffer: 2x2 tiles of 16x16, K=32
        v16bf a0 = load_frag(s_a[cur], wm,      lane);
        v16bf a1 = load_frag(s_a[cur], wm + 16, lane);
        v16bf b0 = load_frag(s_b[cur], wn,      lane);
        v16bf b1 = load_frag(s_b[cur], wn + 16, lane);

        acc00 = __builtin_amdgcn_wmma_f32_16x16x32_bf16(
            false, a0, false, b0, (short)0, acc00, false, false);
        acc01 = __builtin_amdgcn_wmma_f32_16x16x32_bf16(
            false, a0, false, b1, (short)0, acc01, false, false);
        acc10 = __builtin_amdgcn_wmma_f32_16x16x32_bf16(
            false, a1, false, b0, (short)0, acc10, false, false);
        acc11 = __builtin_amdgcn_wmma_f32_16x16x32_bf16(
            false, a1, false, b1, (short)0, acc11, false, false);

        // convert + store next tile into the other buffer (no barrier needed:
        // every wave already passed this iteration's barrier, and only reads
        // buf[cur] until the next one)
        if (more)
            stage_store(s_a[nxt], s_b[nxt], tid, rg);
    }

    // ---- epilogue: C/D layout: VGPR r -> M = r + 8*(lane>=16), N = lane&15
    const int mbase = m0 + wm + ((lane >> 4) << 3);
    const int nl    = lane & 15;
    v8f* accs[2][2] = { {&acc00, &acc01}, {&acc10, &acc11} };
#pragma unroll
    for (int tm = 0; tm < 2; ++tm) {
#pragma unroll
        for (int tn = 0; tn < 2; ++tn) {
            int col = n0 + wn + (tn << 4) + nl;
            if (col >= N) continue;
            float bi = bias ? bias[col] : 0.0f;
            v8f a = *accs[tm][tn];
#pragma unroll
            for (int r = 0; r < 8; ++r) {
                int row = mbase + (tm << 4) + r;
                size_t o = (size_t)row * ldc + col;
                float val = a[r] + bi;
                if (acc_flag) val += C[o];
                C[o] = val;
            }
        }
    }
}

// ---------------------------------------------------------------------------
// Embedding gather: lstm_in[b, 2048 + e] = emb[input_c[b], e]
// ---------------------------------------------------------------------------
__global__ void gather_kernel(const int* __restrict__ input_c,
                              const float* __restrict__ emb,
                              float* __restrict__ lstm_in) {
    int b = blockIdx.x;          // 0..255
    int e = threadIdx.x;         // 0..255
    int idx = input_c[b];
    lstm_in[(size_t)b * 2304 + 2048 + e] = emb[(size_t)idx * 256 + e];
}

// ---------------------------------------------------------------------------
// Fused attention: scores -> softmax(T=64) -> context into lstm_in[:, :2048]
// One 256-thread block per batch row.
// ---------------------------------------------------------------------------
__global__ __launch_bounds__(256)
void attn_kernel(const float* __restrict__ q,      // [256,512]
                 const float* __restrict__ kbuf,   // [256,64,512]
                 const float* __restrict__ vvec,   // [512]
                 const float* __restrict__ bv,     // [1]
                 const float* __restrict__ enc,    // [256,64,2048]
                 float* __restrict__ lstm_in) {    // [256,2304]
    __shared__ float s_q[512];
    __shared__ float s_attn[64];

    const int b    = blockIdx.x;
    const int tid  = threadIdx.x;
    const int wave = tid >> 5;
    const int lane = tid & 31;

    s_q[tid]       = q[(size_t)b * 512 + tid];
    s_q[tid + 256] = q[(size_t)b * 512 + tid + 256];
    __syncthreads();

    // scores: wave w handles t = 8w .. 8w+7
#pragma unroll
    for (int t8 = 0; t8 < 8; ++t8) {
        int t = (wave << 3) + t8;
        const float* kr = &kbuf[((size_t)b * 64 + t) * 512];
        float partial = 0.0f;
        for (int a = lane; a < 512; a += 32)
            partial += tanhf(s_q[a] + kr[a]) * vvec[a];
#pragma unroll
        for (int off = 16; off > 0; off >>= 1)
            partial += __shfl_down(partial, off, 32);
        if (lane == 0) s_attn[t] = partial + bv[0];
    }
    __syncthreads();

    // softmax over 64 by wave 0 (2 elements per lane)
    if (tid < 32) {
        float s0 = s_attn[tid], s1 = s_attn[tid + 32];
        float m = fmaxf(s0, s1);
#pragma unroll
        for (int off = 16; off > 0; off >>= 1)
            m = fmaxf(m, __shfl_down(m, off, 32));
        m = __shfl(m, 0, 32);
        float e0 = __expf(s0 - m), e1 = __expf(s1 - m);
        float s = e0 + e1;
#pragma unroll
        for (int off = 16; off > 0; off >>= 1)
            s += __shfl_down(s, off, 32);
        s = __shfl(s, 0, 32);
        float inv = 1.0f / s;
        s_attn[tid]      = e0 * inv;
        s_attn[tid + 32] = e1 * inv;
    }
    __syncthreads();

    // context[b, e] = sum_t attn[t] * enc[b, t, e]
    for (int e = tid; e < 2048; e += 256) {
        float accv = 0.0f;
        const float* er = &enc[(size_t)b * 64 * 2048 + e];
#pragma unroll 8
        for (int t = 0; t < 64; ++t)
            accv += s_attn[t] * er[(size_t)t * 2048];
        lstm_in[(size_t)b * 2304 + e] = accv;
    }
}

// ---------------------------------------------------------------------------
// LSTM gates: z[256,2048] (i|f|g|o blocks of 512) + carry -> h_new, c_new
// ---------------------------------------------------------------------------
__device__ __forceinline__ float sigmoidf(float x) {
    return 1.0f / (1.0f + __expf(-x));
}

__global__ void gates_kernel(const float* __restrict__ z,
                             const float* __restrict__ carry,
                             float* __restrict__ h_new,
                             float* __restrict__ c_new) {
    int tid = blockIdx.x * blockDim.x + threadIdx.x;   // 0..131071
    int b = tid >> 9;
    int j = tid & 511;
    const float* zr = &z[(size_t)b * 2048];
    float zi = zr[j], zf = zr[512 + j], zg = zr[1024 + j], zo = zr[1536 + j];
    float c = sigmoidf(zf) * carry[tid] + sigmoidf(zi) * tanhf(zg);
    float h = sigmoidf(zo) * tanhf(c);
    h_new[tid] = h;
    c_new[tid] = c;
}

// ---------------------------------------------------------------------------
// Host-side orchestration
// ---------------------------------------------------------------------------
extern "C" void kernel_launch(void* const* d_in, const int* in_sizes, int n_in,
                              void* d_out, int out_size, void* d_ws, size_t ws_size,
                              hipStream_t stream) {
    (void)in_sizes; (void)n_in; (void)out_size; (void)ws_size;

    const int*   input_c = (const int*)  d_in[0];
    const float* hidden  = (const float*)d_in[1];
    const float* carry   = (const float*)d_in[2];
    const float* enc     = (const float*)d_in[3];   // [256,64,2048]
    const float* emb     = (const float*)d_in[4];   // [10000,256]
    const float* w1      = (const float*)d_in[5];   // [512,512]
    const float* b1      = (const float*)d_in[6];
    const float* w2      = (const float*)d_in[7];   // [2048,512]
    const float* b2      = (const float*)d_in[8];
    const float* vvec    = (const float*)d_in[9];   // [512,1]
    const float* bv      = (const float*)d_in[10];  // [1]
    const float* wx      = (const float*)d_in[11];  // [2304,2048]
    const float* wh      = (const float*)d_in[12];  // [512,2048]
    const float* bl      = (const float*)d_in[13];  // [2048]
    const float* wp      = (const float*)d_in[14];  // [512,10000]
    const float* bp      = (const float*)d_in[15];  // [10000]

    // workspace layout (floats)
    float* ws      = (float*)d_ws;
    float* kbuf    = ws;                               // 16384*512 = 8388608
    float* qbuf    = kbuf + 8388608;                   // 256*512   =  131072
    float* lstm_in = qbuf + 131072;                    // 256*2304  =  589824
    float* zbuf    = lstm_in + 589824;                 // 256*2048  =  524288

    // output layout: logits | h_new | c_new
    float* logits = (float*)d_out;
    float* h_new  = logits + 256 * 10000;
    float* c_new  = h_new + 256 * 512;

    // 1) embedding gather into lstm_in[:, 2048:2304]
    gather_kernel<<<256, 256, 0, stream>>>(input_c, emb, lstm_in);

    // 2) q = hidden @ w1 + b1        (256 x 512 x 512)
    gemm_bf16_wmma<<<dim3(512 / 64, 256 / 128), 256, 0, stream>>>(
        hidden, 512, w1, 512, b1, qbuf, 512, 256, 512, 512, 0);

    // 3) k = enc @ w2 + b2           (16384 x 512 x 2048)  -- dominant GEMM
    gemm_bf16_wmma<<<dim3(512 / 64, 16384 / 128), 256, 0, stream>>>(
        enc, 2048, w2, 512, b2, kbuf, 512, 16384, 512, 2048, 0);

    // 4) attention: scores/softmax/context -> lstm_in[:, 0:2048]
    attn_kernel<<<256, 256, 0, stream>>>(qbuf, kbuf, vvec, bv, enc, lstm_in);

    // 5) z = lstm_in @ wx + bl       (256 x 2048 x 2304)
    gemm_bf16_wmma<<<dim3(2048 / 64, 256 / 128), 256, 0, stream>>>(
        lstm_in, 2304, wx, 2048, bl, zbuf, 2048, 256, 2048, 2304, 0);

    // 6) z += hidden @ wh            (256 x 2048 x 512, accumulate)
    gemm_bf16_wmma<<<dim3(2048 / 64, 256 / 128), 256, 0, stream>>>(
        hidden, 512, wh, 2048, nullptr, zbuf, 2048, 256, 2048, 512, 1);

    // 7) gates -> h_new, c_new
    gates_kernel<<<(256 * 512) / 256, 256, 0, stream>>>(zbuf, carry, h_new, c_new);

    // 8) logits = h_new @ wp + bp    (256 x 10000 x 512), N guarded
    gemm_bf16_wmma<<<dim3((10000 + 63) / 64, 256 / 128), 256, 0, stream>>>(
        h_new, 512, wp, 10000, bp, logits, 10000, 256, 10000, 512, 0);
}
